// ShiftedWindowAttention_44298292691605
// MI455X (gfx1250) — matrix operand order repaired and checked by reading, not compiled, not run on hardware
//
#include <hip/hip_runtime.h>
#include <hip/hip_bf16.h>

typedef __attribute__((ext_vector_type(16))) _Float16 v16h;
typedef __attribute__((ext_vector_type(8)))  _Float16 v8h;
typedef __attribute__((ext_vector_type(4)))  _Float16 v4h;
typedef __attribute__((ext_vector_type(8)))  float    v8f;
typedef __attribute__((ext_vector_type(4)))  float    v4f;
typedef int v4i32 __attribute__((vector_size(16)));

#define HEADS 8
#define HD    32
#define NTOK  49
#define NPAD  64
#define CDIM  256
#define BWIN  4096
#define NBH   (BWIN * HEADS)          // 32768
#define MROWS (BWIN * NTOK)           // 200704
#define SCALE 0.17677669529663689f    // 32^-0.5

union FragA { v16h v; v8h h8[2]; };

static __device__ __forceinline__ v8f wmma_f16(v16h a, v16h b, v8f c) {
    // D = A(16x32 f16) * B(32x16 f16) + C(16x16 f32)
    return __builtin_amdgcn_wmma_f32_16x16x32_f16(false, a, false, b, (short)0, c,
                                                  false, false);
}

// --- CDNA5 async global->LDS copy (ASYNCcnt tracked), with safe fallback ----
#if defined(__HIP_DEVICE_COMPILE__) && __has_builtin(__builtin_amdgcn_global_load_async_to_lds_b128)
#define USE_ASYNC_LDS 1
#else
#define USE_ASYNC_LDS 0
#endif

static __device__ __forceinline__ void copy16_g2l(const _Float16* g, _Float16* l) {
#if USE_ASYNC_LDS
    __builtin_amdgcn_global_load_async_to_lds_b128(
        (__attribute__((address_space(1))) v4i32*)(unsigned long long)g,
        (__attribute__((address_space(3))) v4i32*)(unsigned int)(unsigned long long)l,
        0, 0);
#else
    *(v8h*)l = *(const v8h*)g;
#endif
}

static __device__ __forceinline__ void lds_async_wait() {
#if USE_ASYNC_LDS
#if __has_builtin(__builtin_amdgcn_s_wait_asynccnt)
    __builtin_amdgcn_s_wait_asynccnt(0);
#else
    asm volatile("s_wait_asynccnt 0x0" ::: "memory");
#endif
#endif
}

// ---------------------------------------------------------------------------
// Kernel 0: comb[w][h][n*49+m] = rpe[rel_index[n*49+m]*8 + h] + mask[w][n*49+m]
// ---------------------------------------------------------------------------
__global__ void comb_kernel(const float* __restrict__ rpe,
                            const float* __restrict__ mask,
                            const int*  __restrict__ rel,
                            float* __restrict__ comb) {
    const int wh = blockIdx.x;            // w*8 + h
    const int w = wh >> 3, h = wh & 7;
    const float* mbase = mask + (size_t)w * (NTOK * NTOK);
    float* cbase = comb + (size_t)wh * (NTOK * NTOK);
    for (int nm = threadIdx.x; nm < NTOK * NTOK; nm += blockDim.x)
        cbase[nm] = rpe[rel[nm] * HEADS + h] + mbase[nm];
}

// ---------------------------------------------------------------------------
// Kernel 1: QKV projection GEMM  Y(r,o) = sum_k x[r,k]*qkv_w[o,k] + b[o]
// Tiled 128x64x32, 256 threads = 8 wave32 waves, wave = 32x32 macro tile.
// Vectorized f32->f16 staging: b128 loads, packed cvt, b64 LDS stores.
// ---------------------------------------------------------------------------
#define BM 128
#define BN 64
#define BK 32

__global__ __launch_bounds__(256) void qkv_gemm(
    const float* __restrict__ x, const float* __restrict__ w,
    const float* __restrict__ bias,
    _Float16* __restrict__ Q, _Float16* __restrict__ K,
    _Float16* __restrict__ Vt)
{
    __shared__ __align__(16) _Float16 As[BM][BK];
    __shared__ __align__(16) _Float16 Bs[BN][BK];
    const int t    = threadIdx.x;
    const int wave = t >> 5, lane = t & 31;
    const int lo   = lane & 15, hi = lane >> 4;
    const int wm   = wave & 3,  wn = wave >> 2;
    const int rBase = blockIdx.x * BM;
    const int oBase = blockIdx.y * BN;

    v8f acc[2][2] = {};
    for (int kt = 0; kt < CDIM; kt += BK) {
        // prefetch next K-step's 128B lines
        if (kt + BK < CDIM) {
            if (t < BM) __builtin_prefetch(x + (size_t)(rBase + t) * CDIM + kt + BK, 0, 3);
            if (t < BN) __builtin_prefetch(w + (size_t)(oBase + t) * CDIM + kt + BK, 0, 3);
        }
#pragma unroll
        for (int it = 0; it < 4; ++it) {                // A: 128x32 f32 -> f16
            int idx = it * 256 + t;
            int row = idx >> 3, c4 = idx & 7;
            v4f va = *(const v4f*)(x + (size_t)(rBase + row) * CDIM + kt + c4 * 4);
            *(v4h*)&As[row][c4 * 4] = __builtin_convertvector(va, v4h);
        }
#pragma unroll
        for (int it = 0; it < 2; ++it) {                // B: 64x32 f32 -> f16
            int idx = it * 256 + t;
            int o = idx >> 3, c4 = idx & 7;
            v4f vb = *(const v4f*)(w + (size_t)(oBase + o) * CDIM + kt + c4 * 4);
            *(v4h*)&Bs[o][c4 * 4] = __builtin_convertvector(vb, v4h);
        }
        __syncthreads();
        FragA a[2], b[2];
#pragma unroll
        for (int ti = 0; ti < 2; ++ti) {
            const _Float16* ap = &As[wm * 32 + ti * 16 + lo][0];
            a[ti].h8[0] = *(const v8h*)(ap + 8 * hi);
            a[ti].h8[1] = *(const v8h*)(ap + 16 + 8 * hi);
        }
#pragma unroll
        for (int tj = 0; tj < 2; ++tj) {
            const _Float16* bp = &Bs[wn * 32 + tj * 16 + lo][0];
            b[tj].h8[0] = *(const v8h*)(bp + 8 * hi);
            b[tj].h8[1] = *(const v8h*)(bp + 16 + 8 * hi);
        }
#pragma unroll
        for (int ti = 0; ti < 2; ++ti)
#pragma unroll
            for (int tj = 0; tj < 2; ++tj)
                acc[ti][tj] = wmma_f16(a[ti].v, b[tj].v, acc[ti][tj]);
        __syncthreads();
    }
    // Route D fragments into Q / K / Vt layouts
#pragma unroll
    for (int tj = 0; tj < 2; ++tj) {
        const int oG = oBase + wn * 32 + tj * 16 + lo;
        const float bv = bias[oG];
        const int which = oG >> 8;          // 0=q 1=k 2=v
        const int c = oG & 255;
        const int h = c >> 5, d = c & 31;
#pragma unroll
        for (int ti = 0; ti < 2; ++ti) {
#pragma unroll
            for (int r0 = 0; r0 < 8; ++r0) {
                const int r = rBase + wm * 32 + ti * 16 + r0 + 8 * hi;
                const float val = acc[ti][tj][r0] + bv;
                const int bwin = r / NTOK, n = r - bwin * NTOK;
                const size_t bh = (size_t)bwin * HEADS + h;
                if (which == 0)
                    Q[(bh * NTOK + n) * HD + d] = (_Float16)(val * SCALE);
                else if (which == 1)
                    K[(bh * NTOK + n) * HD + d] = (_Float16)val;
                else
                    Vt[(bh * HD + d) * NPAD + n] = (_Float16)val;
            }
        }
    }
}

// ---------------------------------------------------------------------------
// Kernel 2: attention for one (b,h) per block; 4 waves, wave mt owns rows
// n = mt*16 .. mt*16+15.  S = q*k^T (one WMMA per 16x16 tile, K=hd=32),
// softmax(+bias+mask) in LDS using all 32 lanes, ctx = P*V.
// ---------------------------------------------------------------------------
__global__ __launch_bounds__(128) void attn_kernel(
    const _Float16* __restrict__ Q, const _Float16* __restrict__ Kk,
    const _Float16* __restrict__ Vt, const float* __restrict__ comb,
    _Float16* __restrict__ ctx)
{
    __shared__ __align__(16) float    Ssm[4][16][65];   // padded: no bank conflicts
    __shared__ __align__(16) _Float16 Psm[4][16][72];   // 144B rows: 16B aligned
    const int bh = blockIdx.x;
    const int b = bh >> 3, h = bh & 7;
    const int t = threadIdx.x;
    const int mt = t >> 5, lane = t & 31;
    const int lo = lane & 15, hi = lane >> 4;
    const _Float16* qb = Q  + (size_t)bh * NTOK * HD;
    const _Float16* kb = Kk + (size_t)bh * NTOK * HD;
    const _Float16* vb = Vt + (size_t)bh * HD * NPAD;
    const float*    cb = comb + (size_t)((b & 63) * HEADS + h) * (NTOK * NTOK);

    // A fragment: q rows n = mt*16+lo (zeros past 49)
    FragA qa = {};
    {
        const int n = mt * 16 + lo;
        if (n < NTOK) {
            const _Float16* qr = qb + (size_t)n * HD;
            qa.h8[0] = *(const v8h*)(qr + 8 * hi);
            qa.h8[1] = *(const v8h*)(qr + 16 + 8 * hi);
        }
    }
    // S row-block: 4 N-tiles, one WMMA each (K = hd = 32)
#pragma unroll
    for (int nt = 0; nt < 4; ++nt) {
        FragA kf = {};
        const int m = nt * 16 + lo;
        if (m < NTOK) {
            const _Float16* kr = kb + (size_t)m * HD;
            kf.h8[0] = *(const v8h*)(kr + 8 * hi);
            kf.h8[1] = *(const v8h*)(kr + 16 + 8 * hi);
        }
        v8f s = {};
        s = wmma_f16(qa.v, kf.v, s);
#pragma unroll
        for (int r0 = 0; r0 < 8; ++r0)
            Ssm[mt][r0 + 8 * hi][nt * 16 + lo] = s[r0];
    }
    __syncthreads();

    // Softmax: row n = mt*16+lo split between lane pair (lo, lo+16):
    // hi=0 handles cols [0,32), hi=1 handles cols [32,49); shfl_xor combines.
    {
        const int n = mt * 16 + lo;
        float*    srow = &Ssm[mt][lo][0];
        _Float16* prow = &Psm[mt][lo][0];
        const int mstart = hi * 32;
        if (n < NTOK) {
            const float* crow = cb + n * NTOK;
            float vals[32];
            float mx = -1e30f;
#pragma unroll
            for (int i = 0; i < 32; ++i) {
                const int m = mstart + i;
                float v = -1e30f;
                if (m < NTOK) v = srow[m] + crow[m];
                vals[i] = v;
                mx = fmaxf(mx, v);
            }
            mx = fmaxf(mx, __shfl_xor(mx, 16, 32));
            float sum = 0.0f;
#pragma unroll
            for (int i = 0; i < 32; ++i) {
                float e = 0.0f;
                if (mstart + i < NTOK) e = __expf(vals[i] - mx);
                vals[i] = e;
                sum += e;
            }
            sum += __shfl_xor(sum, 16, 32);
            const float rinv = 1.0f / sum;
#pragma unroll
            for (int i = 0; i < 32; ++i)
                prow[mstart + i] = (_Float16)(vals[i] * rinv);
        } else {
#pragma unroll
            for (int i = 0; i < 32; ++i)
                prow[mstart + i] = (_Float16)0.0f;
        }
    }
    __syncthreads();

    // ctx = P (16x64) @ V (64x32): 2 K-steps x 2 d-tiles
    v8f acc2[2] = {};
#pragma unroll
    for (int ks = 0; ks < 2; ++ks) {
        FragA pa;
        const _Float16* pr = &Psm[mt][lo][0];
        pa.h8[0] = *(const v8h*)(pr + ks * 32 + 8 * hi);
        pa.h8[1] = *(const v8h*)(pr + ks * 32 + 16 + 8 * hi);
#pragma unroll
        for (int nt2 = 0; nt2 < 2; ++nt2) {
            FragA vf;
            const _Float16* vr = vb + (size_t)(nt2 * 16 + lo) * NPAD;  // Vt row = d
            vf.h8[0] = *(const v8h*)(vr + ks * 32 + 8 * hi);
            vf.h8[1] = *(const v8h*)(vr + ks * 32 + 16 + 8 * hi);
            acc2[nt2] = wmma_f16(pa.v, vf.v, acc2[nt2]);
        }
    }
#pragma unroll
    for (int nt2 = 0; nt2 < 2; ++nt2) {
        const int d = nt2 * 16 + lo;
#pragma unroll
        for (int r0 = 0; r0 < 8; ++r0) {
            const int n = mt * 16 + r0 + 8 * hi;
            if (n < NTOK)
                ctx[((size_t)b * NTOK + n) * CDIM + h * HD + d] =
                    (_Float16)acc2[nt2][r0];
        }
    }
}

// ---------------------------------------------------------------------------
// Kernel 3: output projection GEMM  out(r,o) = sum_k ctx[r,k]*out_w[o,k] + b[o]
// A (ctx) is already f16: staged via async global->LDS b128 copies.
// ---------------------------------------------------------------------------
__global__ __launch_bounds__(256) void out_gemm(
    const _Float16* __restrict__ ctx, const float* __restrict__ w,
    const float* __restrict__ bias, float* __restrict__ out)
{
    __shared__ __align__(16) _Float16 As[BM][BK];
    __shared__ __align__(16) _Float16 Bs[BN][BK];
    const int t    = threadIdx.x;
    const int wave = t >> 5, lane = t & 31;
    const int lo   = lane & 15, hi = lane >> 4;
    const int wm   = wave & 3,  wn = wave >> 2;
    const int rBase = blockIdx.x * BM;
    const int oBase = blockIdx.y * BN;

    v8f acc[2][2] = {};
    for (int kt = 0; kt < CDIM; kt += BK) {
        if (kt + BK < CDIM) {
            if (t < BM) __builtin_prefetch(ctx + (size_t)(rBase + t) * CDIM + kt + BK, 0, 3);
            if (t < BN) __builtin_prefetch(w + (size_t)(oBase + t) * CDIM + kt + BK, 0, 3);
        }
#pragma unroll
        for (int it = 0; it < 2; ++it) {                // A: 128x32 f16, async
            int idx = it * 256 + t;
            int row = idx >> 2, c8 = idx & 3;
            copy16_g2l(ctx + (size_t)(rBase + row) * CDIM + kt + c8 * 8,
                       &As[row][c8 * 8]);
        }
#pragma unroll
        for (int it = 0; it < 2; ++it) {                // B: 64x32 f32 -> f16
            int idx = it * 256 + t;
            int o = idx >> 3, c4 = idx & 7;
            v4f vb = *(const v4f*)(w + (size_t)(oBase + o) * CDIM + kt + c4 * 4);
            *(v4h*)&Bs[o][c4 * 4] = __builtin_convertvector(vb, v4h);
        }
        lds_async_wait();
        __syncthreads();
        FragA a[2], b[2];
#pragma unroll
        for (int ti = 0; ti < 2; ++ti) {
            const _Float16* ap = &As[wm * 32 + ti * 16 + lo][0];
            a[ti].h8[0] = *(const v8h*)(ap + 8 * hi);
            a[ti].h8[1] = *(const v8h*)(ap + 16 + 8 * hi);
        }
#pragma unroll
        for (int tj = 0; tj < 2; ++tj) {
            const _Float16* bp = &Bs[wn * 32 + tj * 16 + lo][0];
            b[tj].h8[0] = *(const v8h*)(bp + 8 * hi);
            b[tj].h8[1] = *(const v8h*)(bp + 16 + 8 * hi);
        }
#pragma unroll
        for (int ti = 0; ti < 2; ++ti)
#pragma unroll
            for (int tj = 0; tj < 2; ++tj)
                acc[ti][tj] = wmma_f16(a[ti].v, b[tj].v, acc[ti][tj]);
        __syncthreads();
    }
#pragma unroll
    for (int tj = 0; tj < 2; ++tj) {
        const int oG = oBase + wn * 32 + tj * 16 + lo;
        const float bv = bias[oG];
#pragma unroll
        for (int ti = 0; ti < 2; ++ti) {
#pragma unroll
            for (int r0 = 0; r0 < 8; ++r0) {
                const int r = rBase + wm * 32 + ti * 16 + r0 + 8 * hi;
                out[(size_t)r * CDIM + oG] = acc[ti][tj][r0] + bv;
            }
        }
    }
}

// ---------------------------------------------------------------------------
extern "C" void kernel_launch(void* const* d_in, const int* in_sizes, int n_in,
                              void* d_out, int out_size, void* d_ws, size_t ws_size,
                              hipStream_t stream)
{
    (void)in_sizes; (void)n_in; (void)out_size; (void)ws_size;
    const float* x     = (const float*)d_in[0];
    const float* mask  = (const float*)d_in[1];
    const float* qkv_w = (const float*)d_in[2];
    const float* qkv_b = (const float*)d_in[3];
    const float* rpe   = (const float*)d_in[4];
    const float* out_w = (const float*)d_in[5];
    const float* out_b = (const float*)d_in[6];
    const int*   rel   = (const int*)d_in[7];
    float* out = (float*)d_out;

    char* ws = (char*)d_ws;
    size_t off = 0;
    auto wsalloc = [&](size_t bytes) -> char* {
        char* p = ws + off;
        off += (bytes + 255) & ~(size_t)255;
        return p;
    };
    const size_t QK_BYTES  = (size_t)NBH * NTOK * HD * sizeof(_Float16);   // 102.8 MB
    const size_t VT_BYTES  = (size_t)NBH * HD * NPAD * sizeof(_Float16);   // 134.2 MB
    const size_t CTX_BYTES = (size_t)MROWS * CDIM * sizeof(_Float16);      // 102.8 MB
    const size_t CMB_BYTES = (size_t)64 * HEADS * NTOK * NTOK * sizeof(float);

    _Float16* Qh   = (_Float16*)wsalloc(QK_BYTES);
    _Float16* Kh   = (_Float16*)wsalloc(QK_BYTES);
    _Float16* Vth  = (_Float16*)wsalloc(VT_BYTES);
    _Float16* ctxb = (_Float16*)wsalloc(CTX_BYTES);
    float*    comb = (float*)wsalloc(CMB_BYTES);

    (void)hipMemsetAsync(Vth, 0, VT_BYTES, stream);   // zero-pad columns 49..63
    comb_kernel<<<64 * HEADS, 256, 0, stream>>>(rpe, mask, rel, comb);
    qkv_gemm<<<dim3(MROWS / BM, (3 * CDIM) / BN), 256, 0, stream>>>(
        x, qkv_w, qkv_b, Qh, Kh, Vth);
    attn_kernel<<<NBH, 128, 0, stream>>>(Qh, Kh, Vth, comb, ctxb);
    out_gemm<<<dim3(MROWS / BM, CDIM / BN), 256, 0, stream>>>(
        ctxb, out_w, out_b, out);
}
